// Attention_5317169512739
// MI455X (gfx1250) — matrix operand order repaired and checked by reading, not compiled, not run on hardware
//
#include <hip/hip_runtime.h>

typedef __attribute__((ext_vector_type(16))) _Float16 v16h;
typedef __attribute__((ext_vector_type(8)))  _Float16 v8h;
typedef __attribute__((ext_vector_type(4)))  _Float16 v4h;
typedef __attribute__((ext_vector_type(8)))  float    v8f;
typedef __attribute__((ext_vector_type(4)))  float    v4f;
typedef __attribute__((ext_vector_type(4)))  unsigned v4u;
typedef __attribute__((ext_vector_type(8)))  int      v8i;
typedef __attribute__((ext_vector_type(4)))  int      v4i;

#define BQ   2
#define SQ   2048
#define DQ   1024
#define HQ   16
#define HDQ  64
#define NROW (BQ * SQ)   // 4096

#if __has_builtin(__builtin_amdgcn_tensor_load_to_lds)
#define HAVE_TDM 1
#else
#define HAVE_TDM 0
#endif

// ---------------------------------------------------------------------------
// Async global->LDS staging (ASYNCcnt path)
// ---------------------------------------------------------------------------
__device__ __forceinline__ void async_copy_b128(const void* gptr, void* lds_generic) {
  unsigned lo = (unsigned)(unsigned long long)lds_generic;  // LDS byte offset
  asm volatile("global_load_async_to_lds_b128 %0, %1, off"
               :: "v"(lo), "v"(gptr) : "memory");
}
__device__ __forceinline__ void async_wait0() {
#if __has_builtin(__builtin_amdgcn_s_wait_asynccnt)
  __builtin_amdgcn_s_wait_asynccnt(0);
#else
  asm volatile("s_wait_asynccnt 0x0" ::: "memory");
#endif
}

// ---------------------------------------------------------------------------
// Tensor Data Mover: 2D tile (16-bit elements) global -> LDS  (TENSORcnt path)
//  tileX elems/row (contig), tileY rows, row stride strideElems, packed in LDS.
// ---------------------------------------------------------------------------
#if HAVE_TDM
__device__ __forceinline__ void tdm_load_2d(unsigned ldsOff, const void* g,
                                            unsigned tileX, unsigned tileY,
                                            unsigned strideElems) {
  unsigned long long ga = (unsigned long long)g;
  v4u g0;
  g0[0] = 1u;                                    // count=1, user descriptor
  g0[1] = ldsOff;                                // lds_addr (bytes)
  g0[2] = (unsigned)ga;                          // global_addr[31:0]
  g0[3] = (unsigned)(ga >> 32) | 0x80000000u;    // global_addr[56:32] | type=2
  v8i g1;
  g1[0] = (int)(1u << 16);           // data_size=1 (2 bytes); no mask/flags
  g1[1] = (int)(tileX << 16);        // tensor_dim0[15:0] @ bits63:48
  g1[2] = (int)(tileY << 16);        // tensor_dim0 hi=0, tensor_dim1[15:0] @ 95:80
  g1[3] = (int)(tileX << 16);        // tensor_dim1 hi=0, tile_dim0 @ 127:112
  g1[4] = (int)tileY;                // tile_dim1 @ 143:128, tile_dim2=0
  g1[5] = (int)strideElems;          // tensor_dim0_stride[31:0]
  g1[6] = 0;                         // stride hi, tensor_dim1_stride lo
  g1[7] = 0;
  v4i z4 = {0, 0, 0, 0};
#if defined(__clang_major__) && (__clang_major__ >= 23)
  v8i z8 = {0, 0, 0, 0, 0, 0, 0, 0};
  __builtin_amdgcn_tensor_load_to_lds(g0, g1, z4, z4, z8, 0);
#else
  __builtin_amdgcn_tensor_load_to_lds(g0, g1, z4, z4, 0);
#endif
}
__device__ __forceinline__ void tdm_wait0() {
#if __has_builtin(__builtin_amdgcn_s_wait_tensorcnt)
  __builtin_amdgcn_s_wait_tensorcnt((short)0);
#else
  asm volatile("s_wait_tensorcnt 0x0" ::: "memory");
#endif
}
#endif

// ---------------------------------------------------------------------------
// fp32 -> f16 conversion (weights)
// ---------------------------------------------------------------------------
__global__ __launch_bounds__(256) void k_cvt(const float* __restrict__ src,
                                             _Float16* __restrict__ dst, int n) {
  int i = (blockIdx.x * 256 + threadIdx.x) * 4;
  if (i + 3 < n) {
    v4f f = *(const v4f*)(src + i);
    v4h h;
    h[0] = (_Float16)f[0]; h[1] = (_Float16)f[1];
    h[2] = (_Float16)f[2]; h[3] = (_Float16)f[3];
    *(v4h*)(dst + i) = h;
  }
}

// ---------------------------------------------------------------------------
// Projection GEMM: Yh[b,h,s,hd] = sum_d X[b,s,d] * W[o,d],  o = h*64+hd
// 256 thr (8 waves) -> 128 rows x 64 cols. W tile TDM-staged, double-buffered.
// ---------------------------------------------------------------------------
__global__ __launch_bounds__(256) void k_proj(const float* __restrict__ X,
                                              const _Float16* __restrict__ W16,
                                              _Float16* __restrict__ Yh) {
  __shared__ __align__(16) _Float16 Bs[2][64 * 32];  // [buf][col j][k]
  const int tid   = threadIdx.x;
  const int wave  = tid >> 5;
  const int lane  = tid & 31;
  const int lhalf = lane >> 4;
  const int lrow  = lane & 15;
  const int jb = blockIdx.x * 64;
  const int rb = blockIdx.y * 128 + wave * 16;

  const float* aRow = X + (long)(rb + lrow) * DQ;
  const _Float16* wBase = W16 + (long)jb * DQ;   // uniform tile base
#if !HAVE_TDM
  const int stJ = tid >> 2;
  const int stC = (tid & 3) * 8;
#endif

  v8f acc[4] = {};

  // prologue: stage k=0 into buffer 0
#if HAVE_TDM
  if (wave == 0) {
    tdm_load_2d((unsigned)(unsigned long long)&Bs[0][0], wBase, 32, 64, DQ);
    tdm_wait0();
  }
#else
  async_copy_b128(wBase + (long)stJ * DQ + stC, &Bs[0][stJ * 32 + stC]);
  async_wait0();
#endif
  __syncthreads();

  int p = 0;
  for (int k = 0; k < DQ; k += 32) {
    // issue next tile into the other buffer (overlaps with WMMA below)
    if (k + 32 < DQ) {
#if HAVE_TDM
      if (wave == 0)
        tdm_load_2d((unsigned)(unsigned long long)&Bs[p ^ 1][0],
                    wBase + k + 32, 32, 64, DQ);
#else
      async_copy_b128(wBase + (long)stJ * DQ + stC + k + 32,
                      &Bs[p ^ 1][stJ * 32 + stC]);
#endif
    }

    union { v16h v; _Float16 e[16]; } a;
    const float* ap = aRow + k + lhalf * 8;
    v4f f0 = *(const v4f*)(ap);
    v4f f1 = *(const v4f*)(ap + 4);
    v4f f2 = *(const v4f*)(ap + 16);
    v4f f3 = *(const v4f*)(ap + 20);
#pragma unroll
    for (int i = 0; i < 4; ++i) {
      a.e[i]      = (_Float16)f0[i];
      a.e[4 + i]  = (_Float16)f1[i];
      a.e[8 + i]  = (_Float16)f2[i];
      a.e[12 + i] = (_Float16)f3[i];
    }
    __builtin_prefetch(ap + 32, 0, 0);

#pragma unroll
    for (int nt = 0; nt < 4; ++nt) {
      union { v16h v; v8h h[2]; } bf;
      const _Float16* bp = &Bs[p][(nt * 16 + lrow) * 32 + lhalf * 8];
      bf.h[0] = *(const v8h*)bp;
      bf.h[1] = *(const v8h*)(bp + 16);
      acc[nt] = __builtin_amdgcn_wmma_f32_16x16x32_f16(
          false, a.v, false, bf.v, (short)0, acc[nt], false, false);
    }

    // complete the in-flight copy, then flip buffers
#if HAVE_TDM
    if (wave == 0) tdm_wait0();
#else
    async_wait0();
#endif
    __syncthreads();
    p ^= 1;
  }

#pragma unroll
  for (int nt = 0; nt < 4; ++nt) {
    union { v8f v; float f[8]; } r; r.v = acc[nt];
    const int o = jb + nt * 16 + lrow;
    const int h = o >> 6, hd = o & 63;
#pragma unroll
    for (int v = 0; v < 8; ++v) {
      const int row = rb + v + lhalf * 8;
      const int b = row >> 11, s = row & (SQ - 1);
      Yh[(((long)b * HQ + h) * SQ + s) * HDQ + hd] = (_Float16)r.f[v];
    }
  }
}

// ---------------------------------------------------------------------------
// Fused attention (two-pass flash):
//   pass 1: running row max m / sum l of exp(QK^T/8 + mask)
//   pass 2: recompute scores, write normalized weights to OW, accumulate P.V
// Block: 256 thr (8 waves) = 128 q-rows of one (b,h). Loops over all 2048 keys.
// ---------------------------------------------------------------------------
__global__ __launch_bounds__(256) void k_attn(const _Float16* __restrict__ Q16,
                                              const _Float16* __restrict__ K16,
                                              const _Float16* __restrict__ V16,
                                              const float* __restrict__ mask,
                                              float* __restrict__ OW,
                                              _Float16* __restrict__ ctx16) {
  __shared__ __align__(16) _Float16 Pbuf[8 * 16 * 64];  // per-wave P tile (A-layout src)
  __shared__ __align__(16) _Float16 Vt[64 * 64];        // transposed V chunk [d][k]

  const int tid   = threadIdx.x;
  const int wave  = tid >> 5;
  const int lane  = tid & 31;
  const int lhalf = lane >> 4;
  const int lrow  = lane & 15;
  const int qb = blockIdx.x * 128 + wave * 16;   // this wave's 16 q-rows
  const int bh = blockIdx.y;
  const int b = bh >> 4, h = bh & 15;

  const long qkBase = (long)bh * SQ;

  // Q fragments for this wave: 16 rows x 64 d, kept in VGPRs for both passes.
  union AF { v16h v; v8h hh[2]; _Float16 e[16]; };
  AF aq[2];
  {
    const _Float16* qRow = Q16 + (qkBase + qb + lrow) * HDQ;
#pragma unroll
    for (int t = 0; t < 2; ++t) {
      aq[t].hh[0] = *(const v8h*)(qRow + t * 32 + lhalf * 8);
      aq[t].hh[1] = *(const v8h*)(qRow + t * 32 + 16 + lhalf * 8);
    }
  }

  float rm[8], rl[8];
#pragma unroll
  for (int v = 0; v < 8; ++v) { rm[v] = -3.402823466e+38f; rl[v] = 0.f; }

  // ---------------- pass 1: row statistics ----------------
  for (int kb = 0; kb < SQ; kb += 64) {
    v8f sc[4] = {};
#pragma unroll
    for (int t = 0; t < 2; ++t) {
#pragma unroll
      for (int nt = 0; nt < 4; ++nt) {
        const _Float16* kR =
            K16 + (qkBase + kb + nt * 16 + lrow) * HDQ + t * 32 + lhalf * 8;
        union { v16h v; v8h h[2]; } bf;
        bf.h[0] = *(const v8h*)kR;
        bf.h[1] = *(const v8h*)(kR + 16);
        sc[nt] = __builtin_amdgcn_wmma_f32_16x16x32_f16(
            false, aq[t].v, false, bf.v, (short)0, sc[nt], false, false);
      }
    }
    union SM { v8f v; float f[8]; } sm[4];
#pragma unroll
    for (int nt = 0; nt < 4; ++nt) {
      sm[nt].v = sc[nt];
      const int kc = kb + nt * 16 + lrow;
#pragma unroll
      for (int v = 0; v < 8; ++v) {
        const int qr = qb + v + lhalf * 8;
        sm[nt].f[v] = sm[nt].f[v] * 0.125f + mask[((long)b * SQ + qr) * SQ + kc];
      }
    }
    // online max/sum update (row spread across a 16-lane half)
#pragma unroll
    for (int v = 0; v < 8; ++v) {
      float cm = fmaxf(fmaxf(sm[0].f[v], sm[1].f[v]),
                       fmaxf(sm[2].f[v], sm[3].f[v]));
#pragma unroll
      for (int off = 1; off < 16; off <<= 1)
        cm = fmaxf(cm, __shfl_xor(cm, off, 32));
      const float mnew = fmaxf(rm[v], cm);
      float es = __expf(sm[0].f[v] - mnew) + __expf(sm[1].f[v] - mnew) +
                 __expf(sm[2].f[v] - mnew) + __expf(sm[3].f[v] - mnew);
#pragma unroll
      for (int off = 1; off < 16; off <<= 1)
        es += __shfl_xor(es, off, 32);
      rl[v] = rl[v] * __expf(rm[v] - mnew) + es;
      rm[v] = mnew;
    }
  }

  float rinv[8];
#pragma unroll
  for (int v = 0; v < 8; ++v) rinv[v] = 1.0f / rl[v];

  // ---------------- pass 2: write weights + P.V ----------------
  v8f oacc[4] = {};
  const int sR = tid >> 2;          // V chunk row 0..63
  const int sC = (tid & 3) * 16;    // V col base

  for (int kb = 0; kb < SQ; kb += 64) {
    v8f sc[4] = {};
#pragma unroll
    for (int t = 0; t < 2; ++t) {
#pragma unroll
      for (int nt = 0; nt < 4; ++nt) {
        const _Float16* kR =
            K16 + (qkBase + kb + nt * 16 + lrow) * HDQ + t * 32 + lhalf * 8;
        union { v16h v; v8h h[2]; } bf;
        bf.h[0] = *(const v8h*)kR;
        bf.h[1] = *(const v8h*)(kR + 16);
        sc[nt] = __builtin_amdgcn_wmma_f32_16x16x32_f16(
            false, aq[t].v, false, bf.v, (short)0, sc[nt], false, false);
      }
    }
#pragma unroll
    for (int nt = 0; nt < 4; ++nt) {
      union { v8f v; float f[8]; } r; r.v = sc[nt];
      const int kc = kb + nt * 16 + lrow;
#pragma unroll
      for (int v = 0; v < 8; ++v) {
        const int qr = qb + v + lhalf * 8;
        const float s = r.f[v] * 0.125f + mask[((long)b * SQ + qr) * SQ + kc];
        const float p = __expf(s - rm[v]) * rinv[v];
        OW[(qkBase + qr) * SQ + kc] = p;
        Pbuf[wave * 1024 + (v + 8 * lhalf) * 64 + nt * 16 + lrow] = (_Float16)p;
      }
    }

    // stage transposed V chunk [64 k][64 d] -> Vt[d][k]
    __syncthreads();
    {
      const _Float16* vR = V16 + (qkBase + kb + sR) * HDQ + sC;
      v8h t0 = *(const v8h*)(vR);
      v8h t1 = *(const v8h*)(vR + 8);
#pragma unroll
      for (int i = 0; i < 8; ++i) Vt[(sC + i) * 64 + sR]     = t0[i];
#pragma unroll
      for (int i = 0; i < 8; ++i) Vt[(sC + 8 + i) * 64 + sR] = t1[i];
    }
    __syncthreads();

    // O += P.V  (A from Pbuf, B from Vt)
#pragma unroll
    for (int kk = 0; kk < 64; kk += 32) {
      union { v16h v; v8h h[2]; } a;
      const _Float16* pp = &Pbuf[wave * 1024 + lrow * 64 + kk];
      a.h[0] = *(const v8h*)(pp + lhalf * 8);
      a.h[1] = *(const v8h*)(pp + 16 + lhalf * 8);
#pragma unroll
      for (int nt = 0; nt < 4; ++nt) {
        const _Float16* bp = &Vt[(nt * 16 + lrow) * 64 + kk + lhalf * 8];
        union { v16h v; v8h h[2]; } bf;
        bf.h[0] = *(const v8h*)bp;
        bf.h[1] = *(const v8h*)(bp + 16);
        oacc[nt] = __builtin_amdgcn_wmma_f32_16x16x32_f16(
            false, a.v, false, bf.v, (short)0, oacc[nt], false, false);
      }
    }
  }

  // epilogue: merged-head context in f16
#pragma unroll
  for (int nt = 0; nt < 4; ++nt) {
    union { v8f v; float f[8]; } r; r.v = oacc[nt];
    const int d = nt * 16 + lrow;
#pragma unroll
    for (int v = 0; v < 8; ++v) {
      const int s = qb + v + lhalf * 8;
      ctx16[((long)b * SQ + s) * DQ + h * HDQ + d] = (_Float16)r.f[v];
    }
  }
}

// ---------------------------------------------------------------------------
// Output GEMM: out[n,o] = sum_d ctx16[n,d] * Wo[o,d]  (fp32 out)
// W tile async-staged (ASYNCcnt), double-buffered.
// ---------------------------------------------------------------------------
__global__ __launch_bounds__(256) void k_out(const _Float16* __restrict__ A16,
                                             const _Float16* __restrict__ W16,
                                             float* __restrict__ out) {
  __shared__ __align__(16) _Float16 Bs[2][64 * 32];
  const int tid   = threadIdx.x;
  const int wave  = tid >> 5;
  const int lane  = tid & 31;
  const int lhalf = lane >> 4;
  const int lrow  = lane & 15;
  const int jb = blockIdx.x * 64;
  const int rb = blockIdx.y * 128 + wave * 16;

  const _Float16* aRow = A16 + (long)(rb + lrow) * DQ;
  const int stJ = tid >> 2;
  const int stC = (tid & 3) * 8;
  const _Float16* wRow = W16 + (long)(jb + stJ) * DQ + stC;

  v8f acc[4] = {};

  async_copy_b128(wRow, &Bs[0][stJ * 32 + stC]);
  async_wait0();
  __syncthreads();

  int p = 0;
  for (int k = 0; k < DQ; k += 32) {
    if (k + 32 < DQ)
      async_copy_b128(wRow + k + 32, &Bs[p ^ 1][stJ * 32 + stC]);

    union { v16h v; v8h h[2]; } a;
    a.h[0] = *(const v8h*)(aRow + k + lhalf * 8);
    a.h[1] = *(const v8h*)(aRow + k + 16 + lhalf * 8);
    __builtin_prefetch(aRow + k + 32, 0, 0);

#pragma unroll
    for (int nt = 0; nt < 4; ++nt) {
      union { v16h v; v8h h[2]; } bf;
      const _Float16* bp = &Bs[p][(nt * 16 + lrow) * 32 + lhalf * 8];
      bf.h[0] = *(const v8h*)bp;
      bf.h[1] = *(const v8h*)(bp + 16);
      acc[nt] = __builtin_amdgcn_wmma_f32_16x16x32_f16(
          false, a.v, false, bf.v, (short)0, acc[nt], false, false);
    }

    async_wait0();
    __syncthreads();
    p ^= 1;
  }

#pragma unroll
  for (int nt = 0; nt < 4; ++nt) {
    union { v8f v; float f[8]; } r; r.v = acc[nt];
    const int o = jb + nt * 16 + lrow;
#pragma unroll
    for (int v = 0; v < 8; ++v) {
      const int row = rb + v + lhalf * 8;
      out[(long)row * DQ + o] = r.f[v];
    }
  }
}

// ---------------------------------------------------------------------------
extern "C" void kernel_launch(void* const* d_in, const int* in_sizes, int n_in,
                              void* d_out, int out_size, void* d_ws, size_t ws_size,
                              hipStream_t stream) {
  const float* query = (const float*)d_in[0];
  const float* key   = (const float*)d_in[1];
  const float* value = (const float*)d_in[2];
  const float* mask  = (const float*)d_in[3];
  const float* Wq    = (const float*)d_in[4];
  const float* Wk    = (const float*)d_in[5];
  const float* Wv    = (const float*)d_in[6];
  const float* Wo    = (const float*)d_in[7];

  _Float16* ws = (_Float16*)d_ws;
  const size_t WSZ = (size_t)DQ * DQ;     // 1M halves per weight
  const size_t QSZ = (size_t)NROW * DQ;   // 4M halves per activation
  _Float16* Wq16 = ws;
  _Float16* Wk16 = Wq16 + WSZ;
  _Float16* Wv16 = Wk16 + WSZ;
  _Float16* Wo16 = Wv16 + WSZ;
  _Float16* Q16  = Wo16 + WSZ;
  _Float16* K16  = Q16 + QSZ;
  _Float16* V16  = K16 + QSZ;
  _Float16* C16  = V16 + QSZ;   // total: 40 MB of d_ws

  float* outp = (float*)d_out;          // attn_output [B,S,D]
  float* OW   = outp + QSZ;             // attn_weights [B,H,S,S]

  dim3 blk(256);
  const int cvtGrid = (int)(WSZ / (256 * 4));
  k_cvt<<<cvtGrid, blk, 0, stream>>>(Wq, Wq16, (int)WSZ);
  k_cvt<<<cvtGrid, blk, 0, stream>>>(Wk, Wk16, (int)WSZ);
  k_cvt<<<cvtGrid, blk, 0, stream>>>(Wv, Wv16, (int)WSZ);
  k_cvt<<<cvtGrid, blk, 0, stream>>>(Wo, Wo16, (int)WSZ);

  dim3 gProj(DQ / 64, NROW / 128);
  k_proj<<<gProj, blk, 0, stream>>>(query, Wq16, Q16);
  k_proj<<<gProj, blk, 0, stream>>>(key,   Wk16, K16);
  k_proj<<<gProj, blk, 0, stream>>>(value, Wv16, V16);

  dim3 gA(SQ / 128, BQ * HQ);
  k_attn<<<gA, blk, 0, stream>>>(Q16, K16, V16, mask, OW, C16);

  dim3 gO(DQ / 64, NROW / 128);
  k_out<<<gO, blk, 0, stream>>>(C16, Wo16, outp);
}
